// DarkFeat_14620068675692
// MI455X (gfx1250) — matrix operand order repaired and checked by reading, not compiled, not run on hardware
//
#include <hip/hip_runtime.h>
#include <hip/hip_bf16.h>
#include <cstdint>
#include <cstddef>

typedef __attribute__((ext_vector_type(16))) _Float16 v16h;
typedef __attribute__((ext_vector_type(8)))  _Float16 v8h;
typedef __attribute__((ext_vector_type(8)))  float    v8f;

// ---------------- problem constants ----------------
constexpr int H1 = 768, W1 = 1024, C1 = 32;
constexpr int HP1 = H1 + 2, WP1 = W1 + 2;
constexpr int H2 = 384, W2 = 512, C2 = 64;
constexpr int HP2 = H2 + 2, WP2 = W2 + 2;
constexpr int H3 = 192, W3 = 256, C3 = 128;
constexpr int HP3 = H3 + 2, WP3 = W3 + 2;
constexpr int K_TOP = 5000;
constexpr int NSORT = 1 << 20;          // 1,048,576 >= H1*W1 (786,432)
constexpr float EDGE_THR = (10.0f + 1.0f) * (10.0f + 1.0f) / 10.0f; // 12.1

__device__ __forceinline__ int imin(int a, int b) { return a < b ? a : b; }
__device__ __forceinline__ int imax(int a, int b) { return a > b ? a : b; }
__device__ __forceinline__ float softplusf(float x) {
  // stable: max(x,0) + log1p(exp(-|x|))
  return fmaxf(x, 0.0f) + log1pf(expf(-fabsf(x)));
}

// ---------------- utility: zero fill ----------------
__global__ void zero_u32_kernel(uint32_t* __restrict__ p, size_t n) {
  size_t i = (size_t)blockIdx.x * blockDim.x + threadIdx.x;
  size_t st = (size_t)gridDim.x * blockDim.x;
  for (; i < n; i += st) p[i] = 0u;
}

// ---------------- image -> padded NHWC f16 (C=32, ch>=3 zero) ----------------
__global__ void img_pack_kernel(const float* __restrict__ img, _Float16* __restrict__ dst) {
  int i = blockIdx.x * 256 + threadIdx.x;
  if (i >= H1 * W1) return;
  int y = i / W1, x = i - y * W1;
  unsigned o = ((unsigned)(y + 1) * WP1 + (unsigned)(x + 1)) * 32u;
#pragma unroll
  for (int c = 0; c < 3; ++c) dst[o + c] = (_Float16)img[(size_t)c * H1 * W1 + i];
#pragma unroll
  for (int c = 3; c < 32; ++c) dst[o + c] = (_Float16)0.0f;
}

// ---------------- weight packing into WMMA A-fragment layout ----------------
// A-fragment (16x32 f16), code's own contiguous k order (B matches): lane<16 holds M=lane, halfs e: K = e
//                         lane>=16 holds M=lane-16, K = e + 16.
// Apack[((ct*nkc + kc)*32 + lane)*16 + e]
__global__ void pack_weights_kernel(const float* __restrict__ w, _Float16* __restrict__ ap,
                                    int cinReal, int cinPad, int nkc, int total) {
  int idx = blockIdx.x * 256 + threadIdx.x;
  if (idx >= total) return;
  int e    = idx & 15;
  int lane = (idx >> 4) & 31;
  int kc   = (idx >> 9) % nkc;
  int ct   = idx / (nkc * 512);
  int M = lane & 15;
  int K = e + ((lane >= 16) ? 16 : 0);
  int kk = kc * 32 + K;             // global K index = tap*cinPad + cin
  int tap = kk / cinPad;
  int cin = kk - tap * cinPad;
  float v = 0.0f;
  if (cin < cinReal) v = w[((size_t)(ct * 16 + M) * cinReal + cin) * 9 + tap]; // OIHW
  ap[idx] = (_Float16)v;
}

// ---------------- 3x3 conv as implicit GEMM via V_WMMA_F32_16X16X32_F16 ------
// Register-blocked: each wave computes 32 cout x 64 px = 2x4 tiles of 16x16,
// sharing A across pixel tiles and B across cout tiles (8 WMMA / 6 frag loads).
// Addressing decomposed into (uniform scalar row/tap offset) + (k-loop-invariant
// 32-bit lane offset) + (immediate) so loads lower to saddr+voffset+imm form.
// grid: (Wo/256, Ho, Cout/32); block: 128 = 4 waves.
template <int CSH, int STRIDE>
__global__ __launch_bounds__(128)
void conv3x3_wmma_kernel(const _Float16* __restrict__ in, const _Float16* __restrict__ ap,
                         _Float16* __restrict__ out,
                         int wpIn, int cin, int wpOut, int cout) {
  constexpr int CCH = 1 << CSH;                 // cin chunks of 32 per tap
  constexpr int NKC = 9 * CCH;
  const int lane = threadIdx.x & 31;
  const int wave = threadIdx.x >> 5;
  const int tileX = (blockIdx.x * 4 + wave) * 64;   // 64 px per wave
  const int oy = blockIdx.y;
  const int ct0 = blockIdx.z * 2;                   // two cout tiles
  const int n = lane & 15;
  const int sel = lane >> 4;

  v8f acc[2][4];
#pragma unroll
  for (int u = 0; u < 2; ++u)
#pragma unroll
    for (int t = 0; t < 4; ++t) acc[u][t] = {};

  // A fragments: uniform base (SGPR) + per-lane offset (VGPR) + imm per k-chunk
  const unsigned laneA = (unsigned)lane * 16u;
  const _Float16* apw0 = ap + (unsigned)(ct0 * NKC) * 512u;   // uniform
  const _Float16* apw1 = apw0 + (unsigned)NKC * 512u;         // uniform

  // B fragments: per-lane offsets, invariant over the entire k loop
  unsigned laneB[4];
#pragma unroll
  for (int t = 0; t < 4; ++t)
    laneB[t] = (unsigned)((tileX + 16 * t + n) * STRIDE) * (unsigned)cin +
               (unsigned)(sel * 16);

  const unsigned rowStride = (unsigned)wpIn * (unsigned)cin;

  for (int tap = 0; tap < 9; ++tap) {
    const int ky = tap / 3, kx = tap - ky * 3;
    // wave-uniform: lives in SGPRs, one scalar add per tap
    const unsigned rowOff = (unsigned)(oy * STRIDE + ky) * rowStride +
                            (unsigned)kx * (unsigned)cin;
    const _Float16* rowPtr = in + rowOff;
#pragma unroll
    for (int sub = 0; sub < CCH; ++sub) {
      const unsigned kOff = (unsigned)(tap * CCH + sub) * 512u;   // folds to imm
      v16h a0 = *(const v16h*)(apw0 + kOff + laneA);
      v16h a1 = *(const v16h*)(apw1 + kOff + laneA);
#pragma unroll
      for (int t = 0; t < 4; ++t) {
        v16h b = *(const v16h*)(rowPtr + laneB[t] + (unsigned)(sub << 5));
        acc[0][t] = __builtin_amdgcn_wmma_f32_16x16x32_f16(
            false, a0, false, b, (short)0, acc[0][t], false, false);
        acc[1][t] = __builtin_amdgcn_wmma_f32_16x16x32_f16(
            false, a1, false, b, (short)0, acc[1][t], false, false);
      }
    }
  }
  // D layout: VGPR r, lanes<16 -> M=r, lanes>=16 -> M=r+8 ; N = lane&15
  const unsigned outRow = (unsigned)(oy + 1) * (unsigned)wpOut * (unsigned)cout;
#pragma unroll
  for (int u = 0; u < 2; ++u)
#pragma unroll
    for (int t = 0; t < 4; ++t) {
      v8h h;
#pragma unroll
      for (int r = 0; r < 8; ++r) h[r] = (_Float16)acc[u][t][r];
      unsigned o = outRow + (unsigned)(tileX + 16 * t + n + 1) * (unsigned)cout +
                   (unsigned)((ct0 + u) * 16 + sel * 8);
      *(v8h*)(out + o) = h;
    }
}

// ---------------- per-channel mean / rsqrt(var+eps) ----------------
__global__ __launch_bounds__(256)
void bn_stats_kernel(const _Float16* __restrict__ buf, int wp, int C, int Hh, int Ww,
                     float* __restrict__ stats) {
  int c = blockIdx.x;
  __shared__ float s1[256], s2[256];
  float a = 0.0f, b = 0.0f;
  int npix = Hh * Ww;
  for (int i = threadIdx.x; i < npix; i += 256) {
    int y = i / Ww, x = i - y * Ww;
    float v = (float)buf[((size_t)(y + 1) * wp + (x + 1)) * C + c];
    a += v; b += v * v;
  }
  s1[threadIdx.x] = a; s2[threadIdx.x] = b;
  __syncthreads();
  for (int s = 128; s > 0; s >>= 1) {
    if (threadIdx.x < s) { s1[threadIdx.x] += s1[threadIdx.x + s]; s2[threadIdx.x] += s2[threadIdx.x + s]; }
    __syncthreads();
  }
  if (threadIdx.x == 0) {
    float inv = 1.0f / (float)npix;
    float mean = s1[0] * inv;
    float var = s2[0] * inv - mean * mean;
    stats[c] = mean;
    stats[C + c] = rsqrtf(var + 1e-5f);
  }
}

// ---------------- BN / ReLU elementwise ----------------
__global__ void bn_apply_kernel(const _Float16* __restrict__ src, _Float16* __restrict__ dst,
                                const float* __restrict__ stats, int wp, int C, int Hh, int Ww,
                                int useBN, int useRelu) {
  int i = blockIdx.x * 256 + threadIdx.x;
  int n = Hh * Ww * C;
  if (i >= n) return;
  int c = i % C;
  int p = i / C;
  int y = p / Ww, x = p - y * Ww;
  size_t o = ((size_t)(y + 1) * wp + (x + 1)) * C + c;
  float v = (float)src[o];
  if (useBN) v = (v - stats[c]) * stats[C + c];
  if (useRelu) v = fmaxf(v, 0.0f);
  dst[o] = (_Float16)v;
}

// ---------------- peakiness: sv = max_c softplus(x-avg_sp)*softplus(x-avg_ch) --
__global__ void peakiness_kernel(const _Float16* __restrict__ feat, int wp, int C,
                                 int Hh, int Ww, int dil, float* __restrict__ sv) {
  int i = blockIdx.x * 256 + threadIdx.x;
  if (i >= Hh * Ww) return;
  int y = i / Ww, x = i - y * Ww;
  int ysA[3], xsA[3];
#pragma unroll
  for (int t = 0; t < 3; ++t) {
    int yy = y + (t - 1) * dil; if (yy < 0) yy = -yy; if (yy >= Hh) yy = 2 * Hh - 2 - yy;
    int xx = x + (t - 1) * dil; if (xx < 0) xx = -xx; if (xx >= Ww) xx = 2 * Ww - 2 - xx;
    ysA[t] = yy + 1; xsA[t] = xx + 1;   // padded coords
  }
  size_t ctr = ((size_t)(y + 1) * wp + (x + 1)) * C;
  float sum = 0.0f;
  for (int c = 0; c < C; ++c) sum += (float)feat[ctr + c];
  float avgCh = sum / (float)C;
  float best = 0.0f;
  for (int c = 0; c < C; ++c) {
    float v = (float)feat[ctr + c];
    float sp = 0.0f;
#pragma unroll
    for (int ty = 0; ty < 3; ++ty)
#pragma unroll
      for (int tx = 0; tx < 3; ++tx)
        sp += (float)feat[((size_t)ysA[ty] * wp + xsA[tx]) * C + c];
    sp *= (1.0f / 9.0f);
    float al = softplusf(v - sp);
    float be = softplusf(v - avgCh);
    best = fmaxf(best, al * be);
  }
  sv[i] = best;
}

// ---------------- align-corners bilinear + weighted combine ----------------
__device__ __forceinline__ float bilin_ac(const float* __restrict__ m, int h, int w, int y, int x) {
  float fi = (float)y * (float)(h - 1) / (float)(H1 - 1);
  float fj = (float)x * (float)(w - 1) / (float)(W1 - 1);
  int i0 = (int)floorf(fi); int i1 = imin(i0 + 1, h - 1); float wi = fi - (float)i0;
  int j0 = (int)floorf(fj); int j1 = imin(j0 + 1, w - 1); float wj = fj - (float)j0;
  float a = m[(size_t)i0 * w + j0] * (1.0f - wi) + m[(size_t)i1 * w + j0] * wi;
  float b = m[(size_t)i0 * w + j1] * (1.0f - wi) + m[(size_t)i1 * w + j1] * wi;
  return a * (1.0f - wj) + b * wj;
}

__global__ void score_combine_kernel(const float* __restrict__ sv1, const float* __restrict__ sv2,
                                     const float* __restrict__ sv3, float* __restrict__ score) {
  int i = blockIdx.x * 256 + threadIdx.x;
  if (i >= H1 * W1) return;
  int y = i / W1, x = i - y * W1;
  float s = sv1[i] * (1.0f / 6.0f);
  s += (2.0f / 6.0f) * bilin_ac(sv2, H2, W2, y, x);
  s += (3.0f / 6.0f) * bilin_ac(sv3, H3, W3, y, x);
  score[i] = s;
}

// ---------------- mask + sortable key build ----------------
__global__ void mask_keys_kernel(const float* __restrict__ score,
                                 unsigned long long* __restrict__ keys) {
  int i = blockIdx.x * 256 + threadIdx.x;   // grid == NSORT threads exactly
  if (i >= H1 * W1) { keys[i] = 0ULL; return; }
  int y = i / W1, x = i - y * W1;
  float s = score[i];
  bool ok = s > 0.0f;
  // 3x3 NMS (pad -inf == skip OOB)
  float mx = -1e30f;
#pragma unroll
  for (int dy = -1; dy <= 1; ++dy)
#pragma unroll
    for (int dx = -1; dx <= 1; ++dx) {
      int yy = y + dy, xx = x + dx;
      if (yy >= 0 && yy < H1 && xx >= 0 && xx < W1) mx = fmaxf(mx, score[(size_t)yy * W1 + xx]);
    }
  ok = ok && (s == mx);
  ok = ok && (y >= 5 && y < H1 - 5 && x >= 5 && x < W1 - 5);
  // edge mask, dilation 3, zero pad
  auto g = [&](int yy, int xx) -> float {
    return (yy >= 0 && yy < H1 && xx >= 0 && xx < W1) ? score[(size_t)yy * W1 + xx] : 0.0f;
  };
  float dii = g(y - 3, x) - 2.0f * s + g(y + 3, x);
  float djj = g(y, x - 3) - 2.0f * s + g(y, x + 3);
  float dij = 0.25f * (g(y - 3, x - 3) - g(y - 3, x + 3) - g(y + 3, x - 3) + g(y + 3, x + 3));
  float det = dii * djj - dij * dij;
  float tr = dii + djj;
  ok = ok && (det > 0.0f) && (tr * tr <= EDGE_THR * det);
  unsigned u = 0u;
  if (ok) {
    unsigned b = __float_as_uint(s);
    u = (b & 0x80000000u) ? ~b : (b | 0x80000000u);   // monotone float->uint
  }
  keys[i] = ((unsigned long long)u << 32) | (unsigned long long)(0xFFFFFFFFu - (unsigned)i);
}

// ---------------- bitonic sort step (descending overall) ----------------
__global__ void bitonic_step_kernel(unsigned long long* __restrict__ keys, int j, int k) {
  int i = blockIdx.x * 256 + threadIdx.x;   // exactly NSORT threads
  int l = i ^ j;
  if (l <= i) return;
  unsigned long long a = keys[i], b = keys[l];
  bool desc = ((i & k) == 0);
  if (desc ? (a < b) : (a > b)) { keys[i] = b; keys[l] = a; }
}

// ---------------- emit inds + scores ----------------
__global__ void out_kpts_kernel(const unsigned long long* __restrict__ keys,
                                int* __restrict__ inds, float* __restrict__ vals) {
  int t = blockIdx.x * 256 + threadIdx.x;
  if (t >= K_TOP) return;
  unsigned long long kv = keys[t];
  unsigned idx = 0xFFFFFFFFu - (unsigned)kv;
  unsigned u = (unsigned)(kv >> 32);
  float s;
  if (u == 0u) s = -1e30f;
  else s = (u & 0x80000000u) ? __uint_as_float(u ^ 0x80000000u) : __uint_as_float(~u);
  int y = (int)(idx / (unsigned)W1);
  int x = (int)(idx % (unsigned)W1);
  inds[2 * t] = y;
  inds[2 * t + 1] = x;
  vals[t] = s;
}

// ---------------- descriptors: bilinear sample + L2 normalize ----------------
__global__ __launch_bounds__(128)
void out_descs_kernel(const unsigned long long* __restrict__ keys,
                      const _Float16* __restrict__ dm,  // x6_2 padded NHWC (HP3,WP3,128)
                      float* __restrict__ descs) {
  int t = blockIdx.x;
  int c = threadIdx.x;
  __shared__ float red[128];
  unsigned idx = 0xFFFFFFFFu - (unsigned)keys[t];
  int y = (int)(idx / (unsigned)W1);
  int x = (int)(idx % (unsigned)W1);
  float fi = (float)y * 0.25f, fj = (float)x * 0.25f;
  int i0 = imin(imax((int)floorf(fi), 0), H3 - 1);
  int i1 = imin(imax((int)ceilf(fi), 0), H3 - 1);
  int j0 = imin(imax((int)floorf(fj), 0), W3 - 1);
  int j1 = imin(imax((int)ceilf(fj), 0), W3 - 1);
  float di = fi - (float)i0, dj = fj - (float)j0;
  auto f = [&](int yy, int xx) -> float {
    return (float)dm[((size_t)(yy + 1) * WP3 + (xx + 1)) * C3 + c];
  };
  float v = (1.0f - di) * (1.0f - dj) * f(i0, j0) + (1.0f - di) * dj * f(i0, j1) +
            di * (1.0f - dj) * f(i1, j0) + di * dj * f(i1, j1);
  red[c] = v * v;
  __syncthreads();
  for (int s = 64; s > 0; s >>= 1) {
    if (c < s) red[c] += red[c + s];
    __syncthreads();
  }
  float nrm = fmaxf(sqrtf(red[0]), 1e-12f);
  descs[(size_t)t * C3 + c] = v / nrm;
}

// =====================================================================
extern "C" void kernel_launch(void* const* d_in, const int* in_sizes, int n_in,
                              void* d_out, int out_size, void* d_ws, size_t ws_size,
                              hipStream_t stream) {
  (void)in_sizes; (void)n_in; (void)out_size; (void)ws_size;
  const float* image = (const float*)d_in[0];
  const float* w0   = (const float*)d_in[1];
  const float* w1   = (const float*)d_in[2];
  const float* w2   = (const float*)d_in[3];
  const float* w3   = (const float*)d_in[4];
  const float* w4   = (const float*)d_in[5];
  const float* w5   = (const float*)d_in[6];
  const float* w6_0 = (const float*)d_in[7];
  const float* w6_1 = (const float*)d_in[8];
  const float* w6_2 = (const float*)d_in[9];

  // ---- workspace carve (256B aligned) ----
  char* ws = (char*)d_ws;
  size_t off = 0;
  auto carve = [&](size_t bytes) -> void* {
    off = (off + 255) & ~(size_t)255;
    void* p = ws + off;
    off += bytes;
    return p;
  };
  const size_t szA = (size_t)HP1 * WP1 * C1 * 2;   // 768-res f16 padded
  const size_t szC = (size_t)HP2 * WP2 * C2 * 2;   // 384-res
  const size_t szE = (size_t)HP3 * WP3 * C3 * 2;   // 192-res
  _Float16* bufA = (_Float16*)carve(szA);
  _Float16* bufB = (_Float16*)carve(szA);
  _Float16* bufC = (_Float16*)carve(szC);
  _Float16* bufD = (_Float16*)carve(szC);
  _Float16* bufE = (_Float16*)carve(szE);
  _Float16* bufF = (_Float16*)carve(szE);
  size_t actEnd = off;
  _Float16* apack = (_Float16*)carve((size_t)8 * 36 * 512 * 2); // max pack
  float* stats = (float*)carve(256 * sizeof(float));
  float* sv1 = (float*)carve((size_t)H1 * W1 * 4);
  float* sv2 = (float*)carve((size_t)H2 * W2 * 4);
  float* sv3 = (float*)carve((size_t)H3 * W3 * 4);
  float* score = (float*)carve((size_t)H1 * W1 * 4);
  unsigned long long* keys = (unsigned long long*)carve((size_t)NSORT * 8);

  // ---- zero all activation buffers (keeps zero halos valid) ----
  zero_u32_kernel<<<4096, 256, 0, stream>>>((uint32_t*)ws, actEnd / 4);

  // ---- image -> padded NHWC f16 ----
  img_pack_kernel<<<(H1 * W1 + 255) / 256, 256, 0, stream>>>(image, bufA);

  auto runConv = [&](const float* w, _Float16* in, _Float16* out,
                     int cinReal, int cinPad, int cout,
                     int wpIn, int wpOut, int Ho, int Wo, int stride) {
    int nkc = 9 * cinPad / 32;
    int total = (cout / 16) * nkc * 512;
    pack_weights_kernel<<<(total + 255) / 256, 256, 0, stream>>>(w, apack, cinReal, cinPad, nkc, total);
    dim3 g(Wo / 256, Ho, cout / 32);
    if (cinPad == 32 && stride == 1)
      conv3x3_wmma_kernel<0, 1><<<g, 128, 0, stream>>>(in, apack, out, wpIn, cinPad, wpOut, cout);
    else if (cinPad == 32)
      conv3x3_wmma_kernel<0, 2><<<g, 128, 0, stream>>>(in, apack, out, wpIn, cinPad, wpOut, cout);
    else if (cinPad == 64 && stride == 1)
      conv3x3_wmma_kernel<1, 1><<<g, 128, 0, stream>>>(in, apack, out, wpIn, cinPad, wpOut, cout);
    else if (cinPad == 64)
      conv3x3_wmma_kernel<1, 2><<<g, 128, 0, stream>>>(in, apack, out, wpIn, cinPad, wpOut, cout);
    else
      conv3x3_wmma_kernel<2, 1><<<g, 128, 0, stream>>>(in, apack, out, wpIn, cinPad, wpOut, cout);
  };
  auto runStats = [&](_Float16* buf, int wp, int C, int Hh, int Ww) {
    bn_stats_kernel<<<C, 256, 0, stream>>>(buf, wp, C, Hh, Ww, stats);
  };
  auto runBN = [&](_Float16* src, _Float16* dst, int wp, int C, int Hh, int Ww, int bn, int relu) {
    int n = Hh * Ww * C;
    bn_apply_kernel<<<(n + 255) / 256, 256, 0, stream>>>(src, dst, stats, wp, C, Hh, Ww, bn, relu);
  };

  // L0: x0 = relu(bn(conv(img)))            A -> B
  runConv(w0, bufA, bufB, 3, 32, 32, WP1, WP1, H1, W1, 1);
  runStats(bufB, WP1, C1, H1, W1);
  runBN(bufB, bufB, WP1, C1, H1, W1, 1, 1);
  // L1: x1 = relu(conv(x0))                 B -> A ; x1_bn = bn(x1) -> B
  runConv(w1, bufB, bufA, 32, 32, 32, WP1, WP1, H1, W1, 1);
  runBN(bufA, bufA, WP1, C1, H1, W1, 0, 1);          // relu only
  runStats(bufA, WP1, C1, H1, W1);
  runBN(bufA, bufB, WP1, C1, H1, W1, 1, 0);          // x1_bn
  // peakiness scale 1 from x1 (A), dil=3
  peakiness_kernel<<<(H1 * W1 + 255) / 256, 256, 0, stream>>>(bufA, WP1, C1, H1, W1, 3, sv1);
  // L2: x2 = relu(bn(conv(x1_bn, s2)))      B -> C
  runConv(w2, bufB, bufC, 32, 32, 64, WP1, WP2, H2, W2, 2);
  runStats(bufC, WP2, C2, H2, W2);
  runBN(bufC, bufC, WP2, C2, H2, W2, 1, 1);
  // L3: x3 = relu(conv(x2))                 C -> D ; x3_bn -> C
  runConv(w3, bufC, bufD, 64, 64, 64, WP2, WP2, H2, W2, 1);
  runBN(bufD, bufD, WP2, C2, H2, W2, 0, 1);
  runStats(bufD, WP2, C2, H2, W2);
  runBN(bufD, bufC, WP2, C2, H2, W2, 1, 0);
  // peakiness scale 2 from x3 (D), dil=2
  peakiness_kernel<<<(H2 * W2 + 255) / 256, 256, 0, stream>>>(bufD, WP2, C2, H2, W2, 2, sv2);
  // L4: x4 = relu(bn(conv(x3_bn, s2)))      C -> E
  runConv(w4, bufC, bufE, 64, 64, 128, WP2, WP3, H3, W3, 2);
  runStats(bufE, WP3, C3, H3, W3);
  runBN(bufE, bufE, WP3, C3, H3, W3, 1, 1);
  // L5: x5 = relu(bn(conv(x4)))             E -> F
  runConv(w5, bufE, bufF, 128, 128, 128, WP3, WP3, H3, W3, 1);
  runStats(bufF, WP3, C3, H3, W3);
  runBN(bufF, bufF, WP3, C3, H3, W3, 1, 1);
  // L6_0                                    F -> E
  runConv(w6_0, bufF, bufE, 128, 128, 128, WP3, WP3, H3, W3, 1);
  runStats(bufE, WP3, C3, H3, W3);
  runBN(bufE, bufE, WP3, C3, H3, W3, 1, 1);
  // L6_1                                    E -> F
  runConv(w6_1, bufE, bufF, 128, 128, 128, WP3, WP3, H3, W3, 1);
  runStats(bufF, WP3, C3, H3, W3);
  runBN(bufF, bufF, WP3, C3, H3, W3, 1, 1);
  // L6_2: x6_2 = conv(x6_1)  (raw)          F -> E
  runConv(w6_2, bufF, bufE, 128, 128, 128, WP3, WP3, H3, W3, 1);
  // peakiness scale 3 from x6_2 (E), dil=1
  peakiness_kernel<<<(H3 * W3 + 255) / 256, 256, 0, stream>>>(bufE, WP3, C3, H3, W3, 1, sv3);

  // ---- combine score maps ----
  score_combine_kernel<<<(H1 * W1 + 255) / 256, 256, 0, stream>>>(sv1, sv2, sv3, score);

  // ---- mask + keys, bitonic top-K sort ----
  mask_keys_kernel<<<NSORT / 256, 256, 0, stream>>>(score, keys);
  for (int k = 2; k <= NSORT; k <<= 1)
    for (int j = k >> 1; j > 0; j >>= 1)
      bitonic_step_kernel<<<NSORT / 256, 256, 0, stream>>>(keys, j, k);

  // ---- outputs: inds(int32 K*2) | vals(K) | descs(K*128) ----
  int* indsOut = (int*)d_out;
  float* valsOut = (float*)d_out + 2 * K_TOP;
  float* descOut = (float*)d_out + 3 * K_TOP;
  out_kpts_kernel<<<(K_TOP + 255) / 256, 256, 0, stream>>>(keys, indsOut, valsOut);
  out_descs_kernel<<<K_TOP, 128, 0, stream>>>(keys, bufE, descOut);
}
